// PolicyNet_2826088481290
// MI455X (gfx1250) — matrix operand order repaired and checked by reading, not compiled, not run on hardware
//
#include <hip/hip_runtime.h>
#include <hip/hip_bf16.h>
#include <math.h>

// ---------------------------------------------------------------- constants
#define MN      2048      // M nodes == N devices
#define EE      16384
#define DIN     32
#define HIDN    64
#define DM      96
#define NH      4
#define HD      24        // DM / NH
#define FFDIM   256
#define LAY     4
#define NP1     2049
#define SINK_IT 100

typedef __attribute__((ext_vector_type(16))) _Float16 v16h;
typedef __attribute__((ext_vector_type(8)))  float    v8f;
typedef __attribute__((ext_vector_type(4)))  float    v4f;

// ---------------------------------------------------------------- small utils
__global__ void fill_k(float* p, float v, int n) {
  int i = blockIdx.x * blockDim.x + threadIdx.x;
  if (i < n) p[i] = v;
}
__global__ void copy_k(float* dst, const float* src, int n) {
  int i = blockIdx.x * blockDim.x + threadIdx.x;
  if (i < n) dst[i] = src[i];
}

// ---------------------------------------------------------------- GCN pieces
__global__ void deg_count_k(const int* __restrict__ dst, float* deg, int e) {
  int i = blockIdx.x * blockDim.x + threadIdx.x;
  if (i < e) atomicAdd(&deg[dst[i]], 1.0f);
}
__global__ void inv_sqrt_k(const float* deg, float* nrm, int n) {
  int i = blockIdx.x * blockDim.x + threadIdx.x;
  if (i < n) nrm[i] = 1.0f / sqrtf(deg[i]);
}
// agg[dst] += h[src] * nrm[src]   (E real edges, atomics)
__global__ void edge_scatter_k(const int* __restrict__ src, const int* __restrict__ dst,
                               const float* __restrict__ H, const float* __restrict__ nrm,
                               float* agg, int e, int F) {
  int idx = blockIdx.x * blockDim.x + threadIdx.x;
  if (idx >= e * F) return;
  int ei = idx / F, f = idx - ei * F;
  int s = src[ei], d = dst[ei];
  atomicAdd(&agg[(size_t)d * F + f], H[(size_t)s * F + f] * nrm[s]);
}
// self-loop contribution + final row scaling: agg = (agg + h*nrm[i]) * nrm[i]
__global__ void finish_prop_k(const float* __restrict__ H, const float* __restrict__ nrm,
                              float* agg, int m, int F) {
  int idx = blockIdx.x * blockDim.x + threadIdx.x;
  if (idx >= m * F) return;
  int i = idx / F;
  agg[idx] = (agg[idx] + H[idx] * nrm[i]) * nrm[i];
}

// ---------------------------------------------------------------- WMMA GEMM
// D(MxN) = act( A(MxK) * B + bias ) , B either (KxN) row-major (BT=false)
// or given as W(NxK) row-major computing A @ W^T (BT=true).
// ACT: 0 = alpha*dot + bias ; 1 = relu(alpha*dot + bias) ; 2 = |dot| * alpha
// Preconditions used for branch-free loads: Md % 16 == 0, Kd % 8 == 0,
// all base pointers / leading dims multiples of 4 floats (16B-aligned rows).
template <bool BT, int ACT>
__global__ __launch_bounds__(256) void gemm_wmma_f16(
    const float* __restrict__ A, int lda,
    const float* __restrict__ B, int ldb,
    const float* __restrict__ bias,
    float* __restrict__ D, int ldd,
    int Md, int Nd, int Kd, float alpha) {
  const int wave = threadIdx.x >> 5;
  const int lane = threadIdx.x & 31;
  const int g    = lane >> 4;      // half-wave group
  const int l15  = lane & 15;
  const int tm = blockIdx.y * 64 + (wave >> 1) * 16;
  const int tn = blockIdx.x * 32 + (wave & 1) * 16;
  const int arow  = tm + l15;            // always < Md (Md % 16 == 0)
  const int bcol0 = tn + l15;
  const bool  bok   = bcol0 < Nd;
  const int   bcol  = bok ? bcol0 : 0;   // clamped
  const float bmask = bok ? 1.0f : 0.0f;
  const float* __restrict__ Arow = A + (size_t)arow * lda;
  const float* __restrict__ Bbase = BT ? (B + (size_t)bcol * ldb) : (B + bcol);
  v8f acc = {};
  for (int k0 = 0; k0 < Kd; k0 += 32) {
    // ---- A fragment: 16-bit A 16x32 layout -> lane holds K in
    //      [k0+8g, k0+8g+8) and [k0+16+8g, k0+16+8g+8)
    const int kA0 = k0 + g * 8;
    const int kA1 = k0 + 16 + g * 8;
    const float mA0 = (kA0 < Kd) ? 1.0f : 0.0f;   // segments are 8-aligned,
    const float mA1 = (kA1 < Kd) ? 1.0f : 0.0f;   // Kd%8==0 -> all-or-nothing
    v4f a0 = *(const v4f*)(Arow + (kA0 < Kd ? kA0     : 0));
    v4f a1 = *(const v4f*)(Arow + (kA0 < Kd ? kA0 + 4 : 0));
    v4f a2 = *(const v4f*)(Arow + (kA1 < Kd ? kA1     : 0));
    v4f a3 = *(const v4f*)(Arow + (kA1 < Kd ? kA1 + 4 : 0));
    v16h a;
#pragma unroll
    for (int s = 0; s < 4; ++s) {
      a[s]      = (_Float16)(a0[s] * mA0);
      a[4 + s]  = (_Float16)(a1[s] * mA0);
      a[8 + s]  = (_Float16)(a2[s] * mA1);
      a[12 + s] = (_Float16)(a3[s] * mA1);
    }
    // ---- B fragment: lane holds K in [k0+8g, k0+8g+8) and [k0+16+8g, k0+16+8g+8)
    const int kB0 = k0 + g * 8;
    const int kB1 = kB0 + 16;
    const float mB0 = (kB0 < Kd) ? bmask : 0.0f;
    const float mB1 = (kB1 < Kd) ? bmask : 0.0f;
    v16h b;
    if constexpr (BT) {
      v4f b0 = *(const v4f*)(Bbase + (kB0 < Kd ? kB0     : 0));
      v4f b1 = *(const v4f*)(Bbase + (kB0 < Kd ? kB0 + 4 : 0));
      v4f b2 = *(const v4f*)(Bbase + (kB1 < Kd ? kB1     : 0));
      v4f b3 = *(const v4f*)(Bbase + (kB1 < Kd ? kB1 + 4 : 0));
#pragma unroll
      for (int s = 0; s < 4; ++s) {
        b[s]      = (_Float16)(b0[s] * mB0);
        b[4 + s]  = (_Float16)(b1[s] * mB0);
        b[8 + s]  = (_Float16)(b2[s] * mB1);
        b[12 + s] = (_Float16)(b3[s] * mB1);
      }
    } else {
#pragma unroll
      for (int s = 0; s < 8; ++s) {
        int k  = kB0 + s;
        int k2 = kB1 + s;
        float bv  = Bbase[(size_t)(k  < Kd ? k  : 0) * ldb] * mB0;
        float bv2 = Bbase[(size_t)(k2 < Kd ? k2 : 0) * ldb] * mB1;
        b[s]     = (_Float16)bv;
        b[8 + s] = (_Float16)bv2;
      }
    }
    acc = __builtin_amdgcn_wmma_f32_16x16x32_f16(false, a, false, b,
                                                 (short)0, acc, false, false);
    // speculative prefetch of next k-step (global_prefetch_b8; OOB is dropped)
    __builtin_prefetch(Arow + k0 + 32, 0, 1);
    if constexpr (BT) __builtin_prefetch(Bbase + k0 + 32, 0, 1);
  }
  if (bok) {
    float bb = (ACT == 2) ? 0.0f : (bias ? bias[bcol0] : 0.0f);
#pragma unroll
    for (int r = 0; r < 8; ++r) {
      int m = tm + g * 8 + r;      // C/D layout: lanes16-31 hold M = 8+r
      float val = acc[r] * alpha;
      if (ACT == 0)      val = val + bb;
      else if (ACT == 1) val = fmaxf(val + bb, 0.0f);
      else               val = fabsf(val);
      D[(size_t)m * ldd + bcol0] = val;
    }
  }
}

template <bool BT, int ACT>
static void launch_gemm(const float* A, int lda, const float* B, int ldb,
                        const float* bias, float* D, int ldd,
                        int Md, int Nd, int Kd, float alpha, hipStream_t s) {
  dim3 grid((Nd + 31) / 32, (Md + 63) / 64);
  gemm_wmma_f16<BT, ACT><<<grid, 256, 0, s>>>(A, lda, B, ldb, bias, D, ldd,
                                              Md, Nd, Kd, alpha);
}

// ---------------------------------------------------------------- softmax / LN
__global__ __launch_bounds__(256) void softmax_rows_k(float* Ar, int ld, int n) {
  int r = blockIdx.x;
  float* row = Ar + (size_t)r * ld;
  __shared__ float sm[256];
  int t = threadIdx.x;
  float mx = -INFINITY;
  for (int j = t; j < n; j += 256) mx = fmaxf(mx, row[j]);
  sm[t] = mx; __syncthreads();
  for (int s = 128; s > 0; s >>= 1) { if (t < s) sm[t] = fmaxf(sm[t], sm[t + s]); __syncthreads(); }
  mx = sm[0]; __syncthreads();
  float sum = 0.0f;
  for (int j = t; j < n; j += 256) { float e = expf(row[j] - mx); row[j] = e; sum += e; }
  sm[t] = sum; __syncthreads();
  for (int s = 128; s > 0; s >>= 1) { if (t < s) sm[t] += sm[t + s]; __syncthreads(); }
  float inv = 1.0f / sm[0];
  for (int j = t; j < n; j += 256) row[j] *= inv;
}

// t_out = LN(t_in + res) * g + b   (row length DM, block = 128 threads)
__global__ __launch_bounds__(128) void add_ln_k(const float* __restrict__ tin,
                                                const float* __restrict__ res,
                                                const float* __restrict__ g,
                                                const float* __restrict__ b,
                                                float* tout, int Dd) {
  int r = blockIdx.x, j = threadIdx.x;
  __shared__ float sm[128];
  float x = 0.0f;
  if (j < Dd) x = tin[(size_t)r * Dd + j] + res[(size_t)r * Dd + j];
  sm[j] = (j < Dd) ? x : 0.0f; __syncthreads();
  for (int s = 64; s > 0; s >>= 1) { if (j < s) sm[j] += sm[j + s]; __syncthreads(); }
  float mu = sm[0] / Dd; __syncthreads();
  float d = (j < Dd) ? (x - mu) : 0.0f;
  sm[j] = d * d; __syncthreads();
  for (int s = 64; s > 0; s >>= 1) { if (j < s) sm[j] += sm[j + s]; __syncthreads(); }
  float var = sm[0] / Dd;
  if (j < Dd)
    tout[(size_t)r * Dd + j] = (x - mu) / sqrtf(var + 1e-5f) * g[j] + b[j];
}

// ---------------------------------------------------------------- Sinkhorn
__global__ void bin_fill_k(float* Cp, const float* bin, int n, float lamInv) {
  int j = blockIdx.x * blockDim.x + threadIdx.x;
  if (j >= n) return;
  float v = bin[0] * lamInv;
  if (j < n - 1) Cp[(size_t)j * n + (n - 1)] = v;   // last column
  Cp[(size_t)(n - 1) * n + j] = v;                  // last row (incl. corner)
}
__global__ void transpose_k(const float* __restrict__ Cp, float* CpT, int n) {
  int idx = blockIdx.x * blockDim.x + threadIdx.x;
  if (idx >= n * n) return;
  int i = idx / n, j = idx - i * n;
  CpT[(size_t)j * n + i] = Cp[idx];
}
__global__ void fill_logmu_k(float* lm, int n, float nv, float last) {
  int i = blockIdx.x * blockDim.x + threadIdx.x;
  if (i < n) lm[i] = (i == n - 1) ? last : nv;
}
// out[i] = lmu[i] - logsumexp_j( Mrow[i,j] + add[j] )
__global__ __launch_bounds__(256) void lse_phase_k(const float* __restrict__ Mrow,
                                                   const float* __restrict__ add,
                                                   const float* __restrict__ lmu,
                                                   float* outv, int n) {
  int i = blockIdx.x, t = threadIdx.x;
  const float* row = Mrow + (size_t)i * n;
  __shared__ float sm[256];
  float mx = -INFINITY;
  for (int j = t; j < n; j += 256) mx = fmaxf(mx, row[j] + add[j]);
  sm[t] = mx; __syncthreads();
  for (int s = 128; s > 0; s >>= 1) { if (t < s) sm[t] = fmaxf(sm[t], sm[t + s]); __syncthreads(); }
  mx = sm[0]; __syncthreads();
  float sum = 0.0f;
  for (int j = t; j < n; j += 256) sum += expf(row[j] + add[j] - mx);
  sm[t] = sum; __syncthreads();
  for (int s = 128; s > 0; s >>= 1) { if (t < s) sm[t] += sm[t + s]; __syncthreads(); }
  if (t == 0) outv[i] = lmu[i] - (mx + logf(sm[0]));
}
__global__ void scores_k(const float* __restrict__ Cp, const float* __restrict__ u,
                         const float* __restrict__ v, float* out, int n, float plog) {
  int idx = blockIdx.x * blockDim.x + threadIdx.x;
  if (idx >= n * n) return;
  int i = idx / n, j = idx - i * n;
  out[idx] = Cp[idx] + u[i] + v[j] + plog;
}

// ---------------------------------------------------------------- epilogues
__global__ __launch_bounds__(256) void argmax_red_k(const float* __restrict__ S,
                                                    long bstride, long estride, int n,
                                                    float* vout, int* iout) {
  int b = blockIdx.x, t = threadIdx.x;
  const float* p = S + (size_t)b * bstride;
  __shared__ float sv[256];
  __shared__ int   si[256];
  float bv = -INFINITY; int bi = 0;
  for (int j = t; j < n; j += 256) {
    float v = p[(size_t)j * estride];
    if (v > bv) { bv = v; bi = j; }
  }
  sv[t] = bv; si[t] = bi; __syncthreads();
  for (int s = 128; s > 0; s >>= 1) {
    if (t < s) {
      float ov = sv[t + s]; int oi = si[t + s];
      if (ov > sv[t] || (ov == sv[t] && oi < si[t])) { sv[t] = ov; si[t] = oi; }
    }
    __syncthreads();
  }
  if (t == 0) { vout[b] = sv[0]; iout[b] = si[0]; }
}
__global__ __launch_bounds__(256) void entropy_k(const float* __restrict__ S, int ld,
                                                 int n, float* out) {
  int r = blockIdx.x, t = threadIdx.x;
  const float* row = S + (size_t)r * ld;
  __shared__ float sm[256];
  float sum = 0.0f;
  for (int j = t; j < n; j += 256) { float x = row[j]; sum += x * expf(x); }
  sm[t] = sum; __syncthreads();
  for (int s = 128; s > 0; s >>= 1) { if (t < s) sm[t] += sm[t + s]; __syncthreads(); }
  if (t == 0) out[r] = sm[0];
}
__global__ void mutual_k(const float* __restrict__ v0, const int* __restrict__ i0,
                         const int* __restrict__ i1, float* m0out, float* m1out, int n) {
  int i = blockIdx.x * blockDim.x + threadIdx.x;
  if (i >= n) return;
  int a = i0[i];
  m0out[i] = (i1[a] == i) ? expf(v0[i]) : 0.0f;   // mscores0
  int r = i1[i];
  float s1 = 0.0f;
  if (i0[r] == i)                                  // mutual1
    s1 = (i1[i0[r]] == r) ? expf(v0[r]) : 0.0f;    // mscores0[i1[i]]
  m1out[i] = s1;
}

// ---------------------------------------------------------------- driver
static inline int cdiv(int a, int b) { return (a + b - 1) / b; }

extern "C" void kernel_launch(void* const* d_in, const int* in_sizes, int n_in,
                              void* d_out, int out_size, void* d_ws, size_t ws_size,
                              hipStream_t stream) {
  (void)in_sizes; (void)n_in; (void)out_size; (void)ws_size;
  const float* x      = (const float*)d_in[0];
  const int*   edge   = (const int*)  d_in[1];   // [2,E]: src then dst
  const float* devf   = (const float*)d_in[2];
  const float* gw1    = (const float*)d_in[3];
  const float* gb1    = (const float*)d_in[4];
  const float* gw2    = (const float*)d_in[5];
  const float* gb2    = (const float*)d_in[6];
  const float* sa_w   = (const float*)d_in[7];
  const float* sa_b   = (const float*)d_in[8];
  const float* sa_ow  = (const float*)d_in[9];
  const float* sa_ob  = (const float*)d_in[10];
  const float* ca_w   = (const float*)d_in[11];
  const float* ca_b   = (const float*)d_in[12];
  const float* ca_ow  = (const float*)d_in[13];
  const float* ca_ob  = (const float*)d_in[14];
  const float* ln_g   = (const float*)d_in[15];
  const float* ln_b   = (const float*)d_in[16];
  const float* fw1    = (const float*)d_in[17];
  const float* fb1    = (const float*)d_in[18];
  const float* fw2    = (const float*)d_in[19];
  const float* fb2    = (const float*)d_in[20];
  const float* mf_w   = (const float*)d_in[21];
  const float* mf_b   = (const float*)d_in[22];
  const float* bin    = (const float*)d_in[23];

  float* W = (float*)d_ws;
  // workspace layout (floats) -- ~59 MB total
  float* deg   = W;                        // 2048
  float* nrm   = deg   + MN;               // 2048
  float* p1    = nrm   + MN;               // 2048*32
  float* hbuf  = p1    + MN * DIN;         // 2048*64
  float* p2    = hbuf  + MN * HIDN;        // 2048*64
  float* cg    = p2    + MN * HIDN;        // 2048*96
  float* tbuf  = cg    + MN * DM;          // 2048*96
  float* qkv   = tbuf  + MN * DM;          // 2048*288
  float* obuf  = qkv   + MN * 3 * DM;      // 2048*96
  float* tmp   = obuf  + MN * DM;          // 2048*96
  float* tmp2  = tmp   + MN * DM;          // 2048*256
  float* attb  = tmp2  + MN * FFDIM;       // 2048*2048
  float* Cp    = attb  + (size_t)MN * MN;  // 2049*2049
  float* CpT   = Cp    + (size_t)NP1 * NP1;// 2049*2049
  float* uvec  = CpT   + (size_t)NP1 * NP1;// 2049
  float* vvec  = uvec  + NP1;              // 2049
  float* lmu   = vvec  + NP1;              // 2049
  float* v0    = lmu   + NP1;              // 2048
  float* v1    = v0    + MN;               // 2048
  int*   i0    = (int*)(v1 + MN);          // 2048
  int*   i1    = i0    + MN;               // 2048

  float* out    = (float*)d_out;           // scores 2049*2049
  float* outEnt = out + (size_t)NP1 * NP1; // entropy 2048
  float* outM0  = outEnt + MN;             // mscores0 2048
  float* outM1  = outM0 + MN;              // mscores1 2048

  const int*   esrc = edge;
  const int*   edst = edge + EE;
  const float alphaCost = 1.0f / (sqrtf((float)DM) * 0.1f);  // /sqrt(D)/LAM
  const float alphaAtt  = 1.0f / sqrtf((float)HD);
  const float normNeg   = -logf(4096.0f);  // -log(m+n)
  const float lmuLast   = logf(2048.0f) + normNeg;           // log(n)+norm

  // ---------------- GNN: deg / nrm, two propagation hops --------------------
  fill_k<<<cdiv(MN, 256), 256, 0, stream>>>(deg, 1.0f, MN);   // self loops
  deg_count_k<<<cdiv(EE, 256), 256, 0, stream>>>(edst, deg, EE);
  inv_sqrt_k<<<cdiv(MN, 256), 256, 0, stream>>>(deg, nrm, MN);

  fill_k<<<cdiv(MN * DIN, 256), 256, 0, stream>>>(p1, 0.0f, MN * DIN);
  edge_scatter_k<<<cdiv(EE * DIN, 256), 256, 0, stream>>>(esrc, edst, x, nrm, p1, EE, DIN);
  finish_prop_k<<<cdiv(MN * DIN, 256), 256, 0, stream>>>(x, nrm, p1, MN, DIN);
  launch_gemm<false, 0>(p1, DIN, gw1, HIDN, gb1, hbuf, HIDN, MN, HIDN, DIN, 1.0f, stream);

  fill_k<<<cdiv(MN * HIDN, 256), 256, 0, stream>>>(p2, 0.0f, MN * HIDN);
  edge_scatter_k<<<cdiv(EE * HIDN, 256), 256, 0, stream>>>(esrc, edst, hbuf, nrm, p2, EE, HIDN);
  finish_prop_k<<<cdiv(MN * HIDN, 256), 256, 0, stream>>>(hbuf, nrm, p2, MN, HIDN);
  launch_gemm<false, 0>(p2, HIDN, gw2, DM, gb2, cg, DM, MN, DM, HIDN, 1.0f, stream);

  // ---------------- transformer ---------------------------------------------
  copy_k<<<cdiv(MN * DM, 256), 256, 0, stream>>>(tbuf, devf, MN * DM);

  for (int l = 0; l < LAY; ++l) {
    const float* saw = sa_w + (size_t)l * 3 * DM * DM;
    const float* sab = sa_b + (size_t)l * 3 * DM;
    const float* sow = sa_ow + (size_t)l * DM * DM;
    const float* sob = sa_ob + (size_t)l * DM;
    const float* caw = ca_w + (size_t)l * 3 * DM * DM;
    const float* cab = ca_b + (size_t)l * 3 * DM;
    const float* cow = ca_ow + (size_t)l * DM * DM;
    const float* cob = ca_ob + (size_t)l * DM;
    const float* lg  = ln_g + (size_t)l * 3 * DM;
    const float* lb  = ln_b + (size_t)l * 3 * DM;
    const float* w1  = fw1 + (size_t)l * FFDIM * DM;
    const float* b1  = fb1 + (size_t)l * FFDIM;
    const float* w2  = fw2 + (size_t)l * DM * FFDIM;
    const float* b2  = fb2 + (size_t)l * DM;

    // --- self-attention: qkv = t @ sa_w^T + sa_b  (2048 x 288)
    launch_gemm<true, 0>(tbuf, DM, saw, DM, sab, qkv, 3 * DM, MN, 3 * DM, DM, 1.0f, stream);
    for (int h = 0; h < NH; ++h) {
      const float* qh = qkv + h * HD;
      const float* kh = qkv + DM + h * HD;
      const float* vh = qkv + 2 * DM + h * HD;
      launch_gemm<true, 0>(qh, 3 * DM, kh, 3 * DM, nullptr, attb, MN, MN, MN, HD, alphaAtt, stream);
      softmax_rows_k<<<MN, 256, 0, stream>>>(attb, MN, MN);
      launch_gemm<false, 0>(attb, MN, vh, 3 * DM, nullptr, obuf + h * HD, DM, MN, HD, MN, 1.0f, stream);
    }
    launch_gemm<true, 0>(obuf, DM, sow, DM, sob, tmp, DM, MN, DM, DM, 1.0f, stream);
    add_ln_k<<<MN, 128, 0, stream>>>(tbuf, tmp, lg + 0 * DM, lb + 0 * DM, tbuf, DM);

    // --- cross-attention: q from t, k/v from cg
    launch_gemm<true, 0>(tbuf, DM, caw, DM, cab, qkv, 3 * DM, MN, DM, DM, 1.0f, stream);
    launch_gemm<true, 0>(cg, DM, caw + (size_t)DM * DM, DM, cab + DM,
                         qkv + DM, 3 * DM, MN, 2 * DM, DM, 1.0f, stream);
    for (int h = 0; h < NH; ++h) {
      const float* qh = qkv + h * HD;
      const float* kh = qkv + DM + h * HD;
      const float* vh = qkv + 2 * DM + h * HD;
      launch_gemm<true, 0>(qh, 3 * DM, kh, 3 * DM, nullptr, attb, MN, MN, MN, HD, alphaAtt, stream);
      softmax_rows_k<<<MN, 256, 0, stream>>>(attb, MN, MN);
      launch_gemm<false, 0>(attb, MN, vh, 3 * DM, nullptr, obuf + h * HD, DM, MN, HD, MN, 1.0f, stream);
    }
    launch_gemm<true, 0>(obuf, DM, cow, DM, cob, tmp, DM, MN, DM, DM, 1.0f, stream);
    add_ln_k<<<MN, 128, 0, stream>>>(tbuf, tmp, lg + 1 * DM, lb + 1 * DM, tbuf, DM);

    // --- FFN
    launch_gemm<true, 1>(tbuf, DM, w1, DM, b1, tmp2, FFDIM, MN, FFDIM, DM, 1.0f, stream);
    launch_gemm<true, 0>(tmp2, FFDIM, w2, FFDIM, b2, tmp, DM, MN, DM, FFDIM, 1.0f, stream);
    add_ln_k<<<MN, 128, 0, stream>>>(tbuf, tmp, lg + 2 * DM, lb + 2 * DM, tbuf, DM);
  }

  // ---------------- cost -> Cp, Sinkhorn ------------------------------------
  // match = t @ mf_w + mf_b  (reuse obuf)
  launch_gemm<false, 0>(tbuf, DM, mf_w, DM, mf_b, obuf, DM, MN, DM, DM, 1.0f, stream);
  // Cp[0:2048,0:2048] = |cg @ match^T| / sqrt(D) / LAM   (ldd = 2049)
  launch_gemm<true, 2>(cg, DM, obuf, DM, nullptr, Cp, NP1, MN, MN, DM, alphaCost, stream);
  bin_fill_k<<<cdiv(NP1, 256), 256, 0, stream>>>(Cp, bin, NP1, 10.0f);
  transpose_k<<<cdiv(NP1 * NP1, 256), 256, 0, stream>>>(Cp, CpT, NP1);

  fill_k<<<cdiv(NP1, 256), 256, 0, stream>>>(uvec, 0.0f, NP1);
  fill_k<<<cdiv(NP1, 256), 256, 0, stream>>>(vvec, 0.0f, NP1);
  fill_logmu_k<<<cdiv(NP1, 256), 256, 0, stream>>>(lmu, NP1, normNeg, lmuLast);
  for (int it = 0; it < SINK_IT; ++it) {
    lse_phase_k<<<NP1, 256, 0, stream>>>(Cp,  vvec, lmu, uvec, NP1);  // u update
    lse_phase_k<<<NP1, 256, 0, stream>>>(CpT, uvec, lmu, vvec, NP1);  // v update (log_nu==log_mu, m==n)
  }
  scores_k<<<cdiv(NP1 * NP1, 256), 256, 0, stream>>>(Cp, uvec, vvec, out, NP1, -normNeg);

  // ---------------- epilogues -----------------------------------------------
  argmax_red_k<<<MN, 256, 0, stream>>>(out, (long)NP1, 1L,        MN, v0, i0);  // row max
  argmax_red_k<<<MN, 256, 0, stream>>>(out, 1L,        (long)NP1, MN, v1, i1);  // col max
  entropy_k<<<MN, 256, 0, stream>>>(out, NP1, MN, outEnt);
  mutual_k<<<cdiv(MN, 256), 256, 0, stream>>>(v0, i0, i1, outM0, outM1, MN);
}
